// Panoformer_18038862643335
// MI455X (gfx1250) — matrix operand order repaired and checked by reading, not compiled, not run on hardware
//
#include <hip/hip_runtime.h>
#include <hip/hip_bf16.h>

// ---------------------------------------------------------------------------
// Panoformer block for MI455X (gfx1250, wave32, WMMA, async global->LDS,
// double-buffered LDS pipeline).
// B=4, H=64, W=128, C=256, NH=8, K=9, HID=1024, HD=32, L=8192, tokens=32768.
// ---------------------------------------------------------------------------

#define Bv   4
#define Hh   64
#define Ww   128
#define Cc   256
#define NHh  8
#define Kk   9
#define HIDh 1024
#define HDd  32
#define Ll   (Hh * Ww)          // 8192
#define Mtok (Bv * Ll)          // 32768

#define USE_ASYNC_COPY 1

typedef unsigned short ushort_t;
typedef __bf16 bf16_t;
typedef bf16_t v16bf __attribute__((ext_vector_type(16)));
typedef float  v8f   __attribute__((ext_vector_type(8)));
typedef unsigned int u32x4 __attribute__((ext_vector_type(4)));

// ---------------- helpers ----------------

static __device__ __forceinline__ ushort_t f32_to_bf16(float f) {
    union { float f; unsigned u; } v; v.f = f;
    unsigned r = v.u + 0x7FFFu + ((v.u >> 16) & 1u);   // round-to-nearest-even
    return (ushort_t)(r >> 16);
}
static __device__ __forceinline__ float bf16_to_f32(ushort_t h) {
    union { unsigned u; float f; } v; v.u = ((unsigned)h) << 16;
    return v.f;
}
static __device__ __forceinline__ float gelu_f(float x) {
    float x3 = x * x * x;
    return 0.5f * x * (1.0f + tanhf(0.7978845608028654f * (x + 0.044715f * x3)));
}
static __device__ __forceinline__ int imax(int a, int b) { return a > b ? a : b; }
static __device__ __forceinline__ int imin(int a, int b) { return a < b ? a : b; }

// async global->LDS 16B copy (CDNA5 GLOBAL_LOAD_ASYNC_TO_LDS_B128, ASYNCcnt)
static __device__ __forceinline__ void async_copy_b128(void* lds_generic,
                                                       const void* gaddr) {
#if USE_ASYNC_COPY
    unsigned lds32 = (unsigned)(unsigned long long)lds_generic;
    asm volatile("global_load_async_to_lds_b128 %0, %1, off"
                 :
                 : "v"(lds32), "v"(gaddr)
                 : "memory");
#else
    *reinterpret_cast<u32x4*>(lds_generic) =
        *reinterpret_cast<const u32x4*>(gaddr);
#endif
}
static __device__ __forceinline__ void async_copy_wait() {
#if USE_ASYNC_COPY
    asm volatile("s_wait_asynccnt 0x0" ::: "memory");
#endif
}

// ---------------- fp32 -> bf16 convert ----------------

// plain [K x Nreal] -> transposed, padded [Npad x K] bf16 (for GEMM B operand)
__global__ void cvt_padT_kernel(const float* __restrict__ src,
                                ushort_t* __restrict__ dst,
                                int Kd, int Nreal, int Npad) {
    int i = blockIdx.x * blockDim.x + threadIdx.x;
    int n = Npad * Kd;
    if (i >= n) return;
    int nn = i / Kd, k = i - nn * Kd;
    float v = (nn < Nreal) ? src[(size_t)k * Nreal + nn] : 0.0f;
    dst[i] = f32_to_bf16(v);
}

// ---------------- LayerNorm (one row of C=256 per block) ----------------

__global__ void ln_kernel(const float* __restrict__ x,
                          const float* __restrict__ g,
                          const float* __restrict__ b,
                          ushort_t* __restrict__ out) {
    __shared__ float red[Cc];
    __shared__ float stat[2];
    int row = blockIdx.x;
    int t = threadIdx.x;
    float v = x[(size_t)row * Cc + t];
    red[t] = v;
    __syncthreads();
    #pragma unroll
    for (int s = Cc / 2; s > 0; s >>= 1) {
        if (t < s) red[t] += red[t + s];
        __syncthreads();
    }
    if (t == 0) stat[0] = red[0] * (1.0f / Cc);
    __syncthreads();
    float mean = stat[0];
    float d = v - mean;
    red[t] = d * d;
    __syncthreads();
    #pragma unroll
    for (int s = Cc / 2; s > 0; s >>= 1) {
        if (t < s) red[t] += red[t + s];
        __syncthreads();
    }
    if (t == 0) stat[1] = red[0] * (1.0f / Cc);
    __syncthreads();
    float rs = rsqrtf(stat[1] + 1e-5f);
    out[(size_t)row * Cc + t] = f32_to_bf16(d * rs * g[t] + b[t]);
}

// ---------------- WMMA bf16 GEMM: D = A[MxK] * B[KxN] (+bias, epilogue) ----
// B is pre-transposed: BT[Npad x K] row-major.
// Block: 128 threads = 4 waves. Block tile 128(M) x 64(N), K-step 32.
// Each wave computes a 32x64 strip: 2 A-frags x 4 B-frags = 8 WMMAs / K-step.
// Double-buffered LDS: async copies for tile k+1 overlap WMMAs on tile k;
// one barrier + one s_wait_asynccnt per K-step.
// mode 0: f32 store (+ residual if != nullptr)
// mode 1: bf16 store
// mode 2: gelu -> bf16 store

union AFrag { v16bf v; u32x4 q[2]; };

__global__ void __launch_bounds__(128)
gemm_wmma_bf16(const ushort_t* __restrict__ A,
               const ushort_t* __restrict__ BT,
               const float* __restrict__ bias,
               int M, int Nreal, int Npad, int Kd,
               int mode,
               const float* __restrict__ residual,
               float* __restrict__ outF,
               ushort_t* __restrict__ outH) {
    __shared__ ushort_t As[2][128 * 32];   // [buf][128 m][32 k]
    __shared__ ushort_t Bs[2][64 * 32];    // [buf][64 n][32 k] (transposed tile)

    const int t = threadIdx.x;
    const int lane = t & 31;
    const int wv = t >> 5;
    const int tileM = blockIdx.x * 128;
    const int tileN = blockIdx.y * 64;

    v8f acc[8] = {};     // [mi*4 + j]

    const int bRow = t >> 1, bHalf = t & 1;   // B tile: 64 rows x 32 k
    const ushort_t* gA0 = A + (size_t)(tileM + t) * Kd;
    const ushort_t* gB0 = BT + (size_t)(tileN + bRow) * Kd + bHalf * 16;

    // issue one K-tile of async copies into LDS buffer p
    auto issue_tile = [&](int p, int kk) {
        const ushort_t* ga = gA0 + kk;
        ushort_t* la = &As[p][t * 32];
        #pragma unroll
        for (int c = 0; c < 4; ++c)
            async_copy_b128(la + c * 8, ga + c * 8);
        const ushort_t* gb = gB0 + kk;
        ushort_t* lb = &Bs[p][bRow * 32 + bHalf * 16];
        async_copy_b128(lb, gb);
        async_copy_b128(lb + 8, gb + 8);
    };

    issue_tile(0, 0);
    async_copy_wait();
    __syncthreads();

    int p = 0;
    for (int kk = 0; kk < Kd; kk += 32) {
        // kick off next tile into the other buffer (overlaps the WMMAs below)
        if (kk + 32 < Kd) {
            issue_tile(p ^ 1, kk + 32);
            if (kk + 64 < Kd) {
                __builtin_prefetch(gA0 + kk + 64, 0, 1);
                __builtin_prefetch(gB0 + kk + 64, 0, 1);
            }
        }

        // ---- fragments (ISA 7.12.2 lane layouts; contiguous -> ds_load_b128)
        const int khalf = (lane >> 4) * 8;    // A: lanes 16-31 hold K+8
        AFrag af[2];
        #pragma unroll
        for (int mi = 0; mi < 2; ++mi) {
            const int mrow = wv * 32 + mi * 16 + (lane & 15);
            af[mi].q[0] = *reinterpret_cast<const u32x4*>(&As[p][mrow * 32 + khalf]);
            af[mi].q[1] = *reinterpret_cast<const u32x4*>(&As[p][mrow * 32 + khalf + 16]);
        }
        const int nl = lane & 15;
        const int kb = (lane >> 4) * 8;       // B: lanes 16-31 hold K+8
        #pragma unroll
        for (int j = 0; j < 4; ++j) {
            AFrag bfg;
            const int nrow = j * 16 + nl;
            bfg.q[0] = *reinterpret_cast<const u32x4*>(&Bs[p][nrow * 32 + kb]);
            bfg.q[1] = *reinterpret_cast<const u32x4*>(&Bs[p][nrow * 32 + kb + 16]);
            #pragma unroll
            for (int mi = 0; mi < 2; ++mi) {
                acc[mi * 4 + j] = __builtin_amdgcn_wmma_f32_16x16x32_bf16(
                    false, af[mi].v, false, bfg.v, (short)0, acc[mi * 4 + j],
                    false, false);
            }
        }

        // next tile's async writes must have landed, and all waves must be
        // done reading buffer p before it is overwritten next iteration
        async_copy_wait();
        __syncthreads();
        p ^= 1;
    }

    // epilogue: C/D layout — lane n = lane&15, vgpr r -> m = r + (lane>>4)*8
    const int nl = lane & 15;
    #pragma unroll
    for (int mi = 0; mi < 2; ++mi) {
        const int mb = tileM + wv * 32 + mi * 16 + ((lane >> 4) * 8);
        #pragma unroll
        for (int j = 0; j < 4; ++j) {
            int n = tileN + j * 16 + nl;
            float bs = (n < Nreal) ? bias[n] : 0.0f;
            #pragma unroll
            for (int r = 0; r < 8; ++r) {
                int m = mb + r;
                float val = acc[mi * 4 + j][r] + bs;
                size_t oidx = (size_t)m * Npad + n;
                if (mode == 0) {
                    if (residual) val += residual[oidx];
                    outF[oidx] = val;
                } else if (mode == 1) {
                    outH[oidx] = f32_to_bf16(val);
                } else {
                    outH[oidx] = f32_to_bf16(gelu_f(val));
                }
            }
        }
    }
}

// ---------------- deformable attention (sample + softmax-weighted sum) -----
// One wave per (pixel, head); 32 lanes = 32 channels of HD.

__global__ void attn_kernel(const ushort_t* __restrict__ v,     // [M, 256] bf16
                            const float* __restrict__ off,      // [M, 192]
                            const float* __restrict__ att,      // [M, 128]
                            const float* __restrict__ refp,     // [B,H,W,K,2]
                            ushort_t* __restrict__ out) {       // [M, 256] bf16
    int t = threadIdx.x;
    int lane = t & 31;
    int wv = t >> 5;
    int gid = blockIdx.x * 8 + wv;          // B*H*W*NH = 262144
    int nh = gid & 7;
    int bhw = gid >> 3;                     // 0..32767  (b*8192 + h*128 + w)
    int b = bhw >> 13;

    // softmax over K=9 logits
    const float* arow = att + (size_t)bhw * 128 + nh * Kk;
    float lg[Kk];
    float mx = -1e30f;
    #pragma unroll
    for (int k = 0; k < Kk; ++k) { lg[k] = arow[k]; mx = fmaxf(mx, lg[k]); }
    float ssum = 0.0f;
    #pragma unroll
    for (int k = 0; k < Kk; ++k) { lg[k] = __expf(lg[k] - mx); ssum += lg[k]; }
    float inv = 1.0f / ssum;

    const float* orow = off + (size_t)bhw * 192 + nh * (Kk * 2);
    const float* rrow = refp + (size_t)bhw * (Kk * 2);
    const ushort_t* vbase = v + (size_t)b * Ll * Cc + nh * HDd + lane;

    float acc = 0.0f;
    #pragma unroll
    for (int k = 0; k < Kk; ++k) {
        float gx = rrow[k * 2 + 0] + orow[k * 2 + 0];
        float gy = rrow[k * 2 + 1] + orow[k * 2 + 1];
        float px = (gx + 1.0f) * 0.5f * (float)(Ww - 1);
        float py = (gy + 1.0f) * 0.5f * (float)(Hh - 1);
        float x0f = floorf(px), y0f = floorf(py);
        float wx = px - x0f, wy = py - y0f;
        int x0 = (int)x0f, y0 = (int)y0f;
        int x0w = x0 & 127;              // horizontal wrap (power-of-2 W)
        int x1w = (x0 + 1) & 127;
        int y0c = imin(imax(y0, 0), Hh - 1);
        int y1c = imin(imax(y0 + 1, 0), Hh - 1);
        float v00 = bf16_to_f32(vbase[(size_t)(y0c * Ww + x0w) * Cc]);
        float v01 = bf16_to_f32(vbase[(size_t)(y0c * Ww + x1w) * Cc]);
        float v10 = bf16_to_f32(vbase[(size_t)(y1c * Ww + x0w) * Cc]);
        float v11 = bf16_to_f32(vbase[(size_t)(y1c * Ww + x1w) * Cc]);
        float s = v00 * (1.0f - wx) * (1.0f - wy) + v01 * wx * (1.0f - wy)
                + v10 * (1.0f - wx) * wy + v11 * wx * wy;
        acc += lg[k] * inv * s;
    }
    out[(size_t)bhw * Cc + nh * HDd + lane] = f32_to_bf16(acc);
}

// ---------------- depthwise 3x3 conv + GELU (circular W, zero H pad) -------
// One block per pixel; 256 threads loop over 1024 channels.

__global__ void dwconv_kernel(const ushort_t* __restrict__ y,   // [M, 1024] bf16
                              const float* __restrict__ dwk,    // [HID,1,3,3]
                              const float* __restrict__ dwb,    // [HID]
                              ushort_t* __restrict__ y2) {      // [M, 1024] bf16
    int bhw = blockIdx.x;
    int b = bhw >> 13;
    int hw = bhw & 8191;
    int h = hw >> 7;
    int w = hw & 127;
    #pragma unroll
    for (int i = 0; i < HIDh / 256; ++i) {
        int c = threadIdx.x + 256 * i;
        float s = 0.0f;
        #pragma unroll
        for (int dh = 0; dh < 3; ++dh) {
            int hh = h + dh - 1;
            if (hh < 0 || hh >= Hh) continue;
            #pragma unroll
            for (int dw = 0; dw < 3; ++dw) {
                int ww2 = (w + dw - 1) & 127;    // circular width pad
                float xv = bf16_to_f32(
                    y[((size_t)b * Ll + hh * Ww + ww2) * HIDh + c]);
                s += xv * dwk[c * 9 + dh * 3 + dw];
            }
        }
        y2[(size_t)bhw * HIDh + c] = f32_to_bf16(gelu_f(s + dwb[c]));
    }
}

// ---------------------------------------------------------------------------

extern "C" void kernel_launch(void* const* d_in, const int* in_sizes, int n_in,
                              void* d_out, int out_size, void* d_ws, size_t ws_size,
                              hipStream_t stream) {
    (void)in_sizes; (void)n_in; (void)out_size; (void)ws_size;

    const float* x      = (const float*)d_in[0];
    const float* refp   = (const float*)d_in[1];
    const float* ln1_g  = (const float*)d_in[2];
    const float* ln1_b  = (const float*)d_in[3];
    const float* Wv     = (const float*)d_in[4];
    const float* bv     = (const float*)d_in[5];
    const float* Woff   = (const float*)d_in[6];
    const float* boff   = (const float*)d_in[7];
    const float* Wattn  = (const float*)d_in[8];
    const float* battn  = (const float*)d_in[9];
    const float* Wo     = (const float*)d_in[10];
    const float* bo     = (const float*)d_in[11];
    const float* ln2_g  = (const float*)d_in[12];
    const float* ln2_b  = (const float*)d_in[13];
    const float* W1     = (const float*)d_in[14];
    const float* b1     = (const float*)d_in[15];
    const float* dwk    = (const float*)d_in[16];
    const float* dwb    = (const float*)d_in[17];
    const float* W2     = (const float*)d_in[18];
    const float* b2     = (const float*)d_in[19];
    float* out          = (float*)d_out;

    char* ws = (char*)d_ws;
    // workspace layout (bytes)
    const size_t SZ_XN  = (size_t)Mtok * Cc * 2;        // 16 MB   bf16 (xn / xn2)
    const size_t SZ_V   = (size_t)Mtok * Cc * 2;        // 16 MB   bf16
    const size_t SZ_OFF = (size_t)Mtok * 192 * 4;       // 25 MB   f32
    const size_t SZ_ATT = (size_t)Mtok * 128 * 4;       // 16 MB   f32
    const size_t SZ_A   = (size_t)Mtok * Cc * 2;        // 16 MB   bf16
    const size_t SZ_X2  = (size_t)Mtok * Cc * 4;        // 33 MB   f32

    size_t off = 0;
    ushort_t* xn    = (ushort_t*)(ws + off); off += SZ_XN;
    ushort_t* vbuf  = (ushort_t*)(ws + off); size_t O_V = off; off += SZ_V;
    float*    offb  = (float*)(ws + off);    off += SZ_OFF;
    float*    attb  = (float*)(ws + off);    off += SZ_ATT;
    ushort_t* abuf  = (ushort_t*)(ws + off); off += SZ_A;
    float*    x2    = (float*)(ws + off);    off += SZ_X2;
    // pre-transposed bf16 weights: BT[Npad x K]
    ushort_t* WvT   = (ushort_t*)(ws + off); off += (size_t)256 * 256 * 2;
    ushort_t* WoffT = (ushort_t*)(ws + off); off += (size_t)192 * 256 * 2;
    ushort_t* WattT = (ushort_t*)(ws + off); off += (size_t)128 * 256 * 2;
    ushort_t* WoT   = (ushort_t*)(ws + off); off += (size_t)256 * 256 * 2;
    ushort_t* W1T   = (ushort_t*)(ws + off); off += (size_t)1024 * 256 * 2;
    ushort_t* W2T   = (ushort_t*)(ws + off); off += (size_t)256 * 1024 * 2;
    ushort_t* y2buf = (ushort_t*)(ws + off); off += (size_t)Mtok * HIDh * 2;
    // y (gelu(xn2@W1)) aliases the dead v/off/att/a region (75 MB >= 67 MB)
    ushort_t* ybuf  = (ushort_t*)(ws + O_V);
    ushort_t* xn2   = xn;                    // xn dead after GEMM1a/b/c

    auto cvtT = [&](const float* s, ushort_t* d, int Kd, int Nreal, int Npad) {
        int n = Npad * Kd;
        cvt_padT_kernel<<<(n + 255) / 256, 256, 0, stream>>>(s, d, Kd, Nreal, Npad);
    };

    // weights -> bf16, transposed + padded
    cvtT(Wv,    WvT,   256, 256,  256);
    cvtT(Woff,  WoffT, 256, 144,  192);
    cvtT(Wattn, WattT, 256, 72,   128);
    cvtT(Wo,    WoT,   256, 256,  256);
    cvtT(W1,    W1T,   256, 1024, 1024);
    cvtT(W2,    W2T,   1024, 256, 256);

    // LN1: x -> xn (bf16)
    ln_kernel<<<Mtok, Cc, 0, stream>>>(x, ln1_g, ln1_b, xn);

    dim3 blk(128);
    // GEMM1a: v = xn @ Wv + bv   -> bf16 [M,256]
    gemm_wmma_bf16<<<dim3(Mtok / 128, 256 / 64), blk, 0, stream>>>(
        xn, WvT, bv, Mtok, 256, 256, 256, 1, nullptr, nullptr, vbuf);
    // GEMM1b: off = xn @ Woff + boff -> f32 [M,192] (144 real)
    gemm_wmma_bf16<<<dim3(Mtok / 128, 192 / 64), blk, 0, stream>>>(
        xn, WoffT, boff, Mtok, 144, 192, 256, 0, nullptr, offb, nullptr);
    // GEMM1c: att = xn @ Wattn + battn -> f32 [M,128] (72 real)
    gemm_wmma_bf16<<<dim3(Mtok / 128, 128 / 64), blk, 0, stream>>>(
        xn, WattT, battn, Mtok, 72, 128, 256, 0, nullptr, attb, nullptr);

    // deformable attention -> abuf (bf16)
    attn_kernel<<<(Mtok * NHh) / 8, 256, 0, stream>>>(vbuf, offb, attb, refp, abuf);

    // GEMM2: x2 = abuf @ Wo + bo + x  -> f32
    gemm_wmma_bf16<<<dim3(Mtok / 128, 256 / 64), blk, 0, stream>>>(
        abuf, WoT, bo, Mtok, 256, 256, 256, 0, x, x2, nullptr);

    // LN2: x2 -> xn2 (bf16)
    ln_kernel<<<Mtok, Cc, 0, stream>>>(x2, ln2_g, ln2_b, xn2);

    // GEMM3: y = gelu(xn2 @ W1 + b1) -> bf16 [M,1024]
    gemm_wmma_bf16<<<dim3(Mtok / 128, 1024 / 64), blk, 0, stream>>>(
        xn2, W1T, b1, Mtok, 1024, 1024, 256, 2, nullptr, nullptr, ybuf);

    // depthwise 3x3 + gelu -> y2 (bf16)
    dwconv_kernel<<<Mtok, 256, 0, stream>>>(ybuf, dwk, dwb, y2buf);

    // GEMM4: out = y2 @ W2 + b2 + x2 -> f32
    gemm_wmma_bf16<<<dim3(Mtok / 128, 256 / 64), blk, 0, stream>>>(
        y2buf, W2T, b2, Mtok, 256, 256, 1024, 0, x2, out, nullptr);
}
